// Attention_11295763988829
// MI455X (gfx1250) — matrix operand order repaired and checked by reading, not compiled, not run on hardware
//
#include <hip/hip_runtime.h>
#include <math.h>

typedef __attribute__((ext_vector_type(16))) _Float16 v16h;
typedef __attribute__((ext_vector_type(8)))  float    v8f;

constexpr int B = 8;
constexpr int S = 4096;
constexpr int D = 1024;

constexpr int SCH  = 64;          // s-chunks in pass 3
constexpr int SLEN = S / SCH;     // 64

// ---------------------------------------------------------------------------
// Pass 1: scores[b,s] = <data[b,s,:], crit[b,:]> via WMMA f16 with hi/lo split
// One wave handles a 16-row tile; 8 waves / 256-thread block; 256 blocks.
// ---------------------------------------------------------------------------
__device__ inline void split4(float a, float b2, float c2, float d2,
                              v16h& hi, v16h& lo, int base) {
    float xs[4] = {a, b2, c2, d2};
#pragma unroll
    for (int i = 0; i < 4; ++i) {
        _Float16 h = (_Float16)xs[i];
        hi[base + i] = h;
        lo[base + i] = (_Float16)(xs[i] - (float)h);
    }
}

__global__ void __launch_bounds__(256)
scores_kernel(const float* __restrict__ data, const float* __restrict__ crit,
              float* __restrict__ scores) {
    __shared__ __align__(32) _Float16 chi[D];
    __shared__ __align__(32) _Float16 clo[D];

    const int bb = blockIdx.x / 32;            // 32 blocks per batch

    // stage crit[bb] as f16 hi/lo in LDS (one time per block)
    for (int i = threadIdx.x; i < D; i += 256) {
        float x = crit[bb * D + i];
        _Float16 h = (_Float16)x;
        chi[i] = h;
        clo[i] = (_Float16)(x - (float)h);
    }
    __syncthreads();

    const int wave = threadIdx.x >> 5;
    const int lane = threadIdx.x & 31;
    const int half = lane >> 4;                // lane group: 0 or 1
    const int mrow = lane & 15;                // row within 16-row tile

    const int w  = blockIdx.x * 8 + wave;      // global tile id (0..2047)
    const int s0 = (w & 255) * 16;             // tile start row within batch

    // A layout (16-bit A 16x32): group h loads K = k0+16h+0..15
    const float* prow = data + ((size_t)bb * S + s0 + mrow) * (size_t)D + half * 16;

    v8f c = {};

    for (int k0 = 0; k0 < D; k0 += 32) {
        const float4 x0 = *(const float4*)(prow + k0);
        const float4 x1 = *(const float4*)(prow + k0 + 4);
        const float4 x2 = *(const float4*)(prow + k0 + 8);
        const float4 x3 = *(const float4*)(prow + k0 + 12);

        v16h ah, al;
        split4(x0.x, x0.y, x0.z, x0.w, ah, al, 0);
        split4(x1.x, x1.y, x1.z, x1.w, ah, al, 4);
        split4(x2.x, x2.y, x2.z, x2.w, ah, al, 8);
        split4(x3.x, x3.y, x3.z, x3.w, ah, al, 12);

        // B layout (16-bit B 32x16): lane group h holds K = k0+16h+j, all N equal
        const v16h bh = *(const v16h*)(&chi[k0 + half * 16]);
        const v16h bl = *(const v16h*)(&clo[k0 + half * 16]);

        // (ah+al)*(bh+bl) ~= ah*bh + al*bh + ah*bl  (drop al*bl ~ 2^-22)
        c = __builtin_amdgcn_wmma_f32_16x16x32_f16(false, ah, false, bh,
                                                   (short)0, c, false, false);
        c = __builtin_amdgcn_wmma_f32_16x16x32_f16(false, al, false, bh,
                                                   (short)0, c, false, false);
        c = __builtin_amdgcn_wmma_f32_16x16x32_f16(false, ah, false, bl,
                                                   (short)0, c, false, false);
    }

    // C/D layout: lane 0 holds rows 0..7 (c[0..7]); lane 16 holds rows 8..15
    if (mrow == 0) {
        float* sp = scores + bb * S + s0 + half * 8;
#pragma unroll
        for (int r = 0; r < 8; ++r) sp[r] = c[r];
    }
}

// ---------------------------------------------------------------------------
// Pass 2: softmax over S per batch. One 256-thread block per batch.
// ---------------------------------------------------------------------------
__global__ void __launch_bounds__(256)
softmax_kernel(const float* __restrict__ scores, float* __restrict__ weights) {
    __shared__ float red[256];
    const int b   = blockIdx.x;
    const int tid = threadIdx.x;

    float x[16];
    float mx = -INFINITY;
#pragma unroll
    for (int j = 0; j < 16; ++j) {
        x[j] = scores[b * S + tid + j * 256];
        mx = fmaxf(mx, x[j]);
    }
    red[tid] = mx;
    __syncthreads();
    for (int off = 128; off > 0; off >>= 1) {
        if (tid < off) red[tid] = fmaxf(red[tid], red[tid + off]);
        __syncthreads();
    }
    mx = red[0];
    __syncthreads();

    float e[16];
    float lsum = 0.f;
#pragma unroll
    for (int j = 0; j < 16; ++j) {
        e[j] = expf(x[j] - mx);
        lsum += e[j];
    }
    red[tid] = lsum;
    __syncthreads();
    for (int off = 128; off > 0; off >>= 1) {
        if (tid < off) red[tid] += red[tid + off];
        __syncthreads();
    }
    const float inv = 1.0f / red[0];
#pragma unroll
    for (int j = 0; j < 16; ++j)
        weights[b * S + tid + j * 256] = e[j] * inv;
}

// ---------------------------------------------------------------------------
// Pass 3: partial[sch][b][d] = sum_{s in chunk} w[b,s] * data[b,s,d]
// Block = (b, schunk); each thread owns 4 consecutive d (b128 loads, 4KB/s
// per block, fully coalesced); weights for the chunk staged in LDS.
// ---------------------------------------------------------------------------
__global__ void __launch_bounds__(256)
partial_kernel(const float* __restrict__ data, const float* __restrict__ weights,
               float* __restrict__ partial) {
    const int blk = blockIdx.x;                  // B * SCH = 512
    const int b   = blk / SCH;
    const int sch = blk % SCH;

    const int d  = threadIdx.x * 4;              // 256 threads cover D = 1024
    const int s0 = sch * SLEN;

    __shared__ float wsh[SLEN];
    if (threadIdx.x < SLEN)
        wsh[threadIdx.x] = weights[b * S + s0 + threadIdx.x];
    __syncthreads();

    const float* p = data + ((size_t)b * S + s0) * (size_t)D + d;
    float4 acc = {0.f, 0.f, 0.f, 0.f};
#pragma unroll 4
    for (int j = 0; j < SLEN; ++j) {
        const float4 v = *(const float4*)(p + (size_t)j * D);
        const float  wj = wsh[j];
        acc.x = fmaf(wj, v.x, acc.x);
        acc.y = fmaf(wj, v.y, acc.y);
        acc.z = fmaf(wj, v.z, acc.z);
        acc.w = fmaf(wj, v.w, acc.w);
    }

    *(float4*)(partial + ((size_t)sch * B + b) * (size_t)D + d) = acc;
}

// ---------------------------------------------------------------------------
// Pass 4: deterministic fixed-order reduction of the SCH partials (float4).
// ---------------------------------------------------------------------------
__global__ void __launch_bounds__(256)
reduce_kernel(const float* __restrict__ partial, float* __restrict__ out) {
    const int idx = (blockIdx.x * 256 + threadIdx.x) * 4;   // 0 .. B*D-1 step 4
    float4 acc = {0.f, 0.f, 0.f, 0.f};
#pragma unroll
    for (int k = 0; k < SCH; ++k) {
        const float4 v = *(const float4*)(partial + (size_t)k * (B * D) + idx);
        acc.x += v.x; acc.y += v.y; acc.z += v.z; acc.w += v.w;
    }
    *(float4*)(out + idx) = acc;
}

// ---------------------------------------------------------------------------
extern "C" void kernel_launch(void* const* d_in, const int* in_sizes, int n_in,
                              void* d_out, int out_size, void* d_ws, size_t ws_size,
                              hipStream_t stream) {
    const float* data = (const float*)d_in[0];   // [B,S,D] f32
    const float* crit = (const float*)d_in[1];   // [B,D]   f32
    float*       out  = (float*)d_out;           // [B,D]   f32

    float* scores  = (float*)d_ws;               // B*S            = 32768 f
    float* weights = scores + B * S;             // B*S            = 32768 f
    float* partial = weights + B * S;            // SCH*B*D        = 524288 f
    (void)in_sizes; (void)n_in; (void)out_size; (void)ws_size;

    scores_kernel <<<dim3((B * S / 16) / 8), dim3(256), 0, stream>>>(data, crit, scores);
    softmax_kernel<<<dim3(B),                dim3(256), 0, stream>>>(scores, weights);
    partial_kernel<<<dim3(B * SCH),          dim3(256), 0, stream>>>(data, weights, partial);
    reduce_kernel <<<dim3(B * D / 256 / 4),  dim3(256), 0, stream>>>(partial, out);
}
